// SparseTopKMoE_773094113657
// MI455X (gfx1250) — compile-verified, matrix operand order checked by reading, not run on hardware
//
#include <hip/hip_runtime.h>
#include <math.h>

// Problem constants (match the reference).
#define C_DIM   256
#define HID_DIM 512
#define E_NUM   8
#define HW_DIM  4096
#define B_NUM   16
#define N_TOK   (B_NUM * HW_DIM)      /* 65536 tokens */
#define MAXTILE 512                    /* ceil(N_TOK/128) */

// Padded LDS strides (elements) -> bank-conflict-free ds_load_b128 fragments.
#define TOK_STRIDE 264                 /* 256 + 8 pad ; 132 dwords/row      */
#define H_STRIDE   72                  /* 64 + 8 pad  ; 36 dwords/row       */

typedef __attribute__((ext_vector_type(16))) __bf16 v16bf;
typedef __attribute__((ext_vector_type(8)))  float  v8f;

union Frag {
  v16bf bf;
  uint4 q[2];
};

__device__ __forceinline__ unsigned short f2bf(float f) {
  unsigned u = __float_as_uint(f);
  u += 0x7fffu + ((u >> 16) & 1u);      // round-to-nearest-even
  return (unsigned short)(u >> 16);
}
__device__ __forceinline__ float gelu_exact(float x) {
  return 0.5f * x * (1.0f + erff(x * 0.70710678118654752440f));
}

// ---------------------------------------------------------------- zero cursors
__global__ void k_zero(int* cursors) {
  if (threadIdx.x < E_NUM) cursors[threadIdx.x] = 0;
}

// ---------------------------------------------------------------- out = x
__global__ void k_copy(const float4* __restrict__ in, float4* __restrict__ out) {
  int i = blockIdx.x * blockDim.x + threadIdx.x;   // 4,194,304 total
  out[i] = in[i];
}

// ------------------------------------------------- generic transpose + fp32->bf16
// in:  R x Cc row-major fp32 ; out: Cc x R row-major bf16 ; one batch per blockIdx.z
__global__ void k_transpose_bf16(const float* __restrict__ in,
                                 unsigned short* __restrict__ out,
                                 int R, int Cc, long strideIn, long strideOut) {
  __shared__ float tile[32][33];
  in  += (long)blockIdx.z * strideIn;
  out += (long)blockIdx.z * strideOut;
  int c0 = blockIdx.x * 32;        // col in input
  int r0 = blockIdx.y * 32;        // row in input
  int tx = threadIdx.x, ty = threadIdx.y;   // 32 x 8
#pragma unroll
  for (int i = 0; i < 4; ++i) {
    int r = r0 + ty + i * 8;
    tile[ty + i * 8][tx] = in[(long)r * Cc + c0 + tx];
  }
  __syncthreads();
#pragma unroll
  for (int i = 0; i < 4; ++i) {
    int cc = c0 + ty + i * 8;
    out[(long)cc * R + r0 + tx] = f2bf(tile[tx][ty + i * 8]);
  }
}

// --------------------------------------------------- router + top2 + scatter
__global__ void k_router(const unsigned short* __restrict__ tokB,
                         const float* __restrict__ Wr,   // (C,E) row-major
                         const float* __restrict__ br,   // (E)
                         int*   __restrict__ btok,       // [E][N_TOK]
                         float* __restrict__ bwt,        // [E][N_TOK]
                         int*   __restrict__ cursors) {
  __shared__ float sWr[C_DIM * E_NUM];   // 8 KiB
  int tid = threadIdx.x;
  for (int i = tid; i < C_DIM * E_NUM; i += 256) sWr[i] = Wr[i];
  __syncthreads();

  int t = blockIdx.x * 256 + tid;        // < N_TOK exactly
  float acc[E_NUM];
#pragma unroll
  for (int e = 0; e < E_NUM; ++e) acc[e] = br[e];

  const uint4* row = (const uint4*)(tokB + (size_t)t * C_DIM);
#pragma unroll 4
  for (int q = 0; q < 32; ++q) {
    uint4 v = row[q];
    unsigned d[4] = {v.x, v.y, v.z, v.w};
#pragma unroll
    for (int j = 0; j < 4; ++j) {
      int c = q * 8 + j * 2;
      float x0 = __uint_as_float(d[j] << 16);
      float x1 = __uint_as_float(d[j] & 0xffff0000u);
#pragma unroll
      for (int e = 0; e < E_NUM; ++e)
        acc[e] += x0 * sWr[c * E_NUM + e] + x1 * sWr[(c + 1) * E_NUM + e];
    }
  }
  // top-2 (descending, earliest index wins ties — matches lax.top_k)
  int i0 = 0, i1 = 1;
  float v0 = acc[0], v1 = acc[1];
  if (v1 > v0) { float tv = v0; v0 = v1; v1 = tv; i0 = 1; i1 = 0; }
#pragma unroll
  for (int e = 2; e < E_NUM; ++e) {
    float v = acc[e];
    if (v > v0)      { v1 = v0; i1 = i0; v0 = v; i0 = e; }
    else if (v > v1) { v1 = v; i1 = e; }
  }
  float e1 = __expf(v1 - v0);
  float w0 = 1.0f / (1.0f + e1);
  float w1 = e1 * w0;

  int p0 = atomicAdd(&cursors[i0], 1);
  btok[i0 * N_TOK + p0] = t;  bwt[i0 * N_TOK + p0] = w0;
  int p1 = atomicAdd(&cursors[i1], 1);
  btok[i1 * N_TOK + p1] = t;  bwt[i1 * N_TOK + p1] = w1;
}

// --------------------------------------------------- fused expert GEMM (WMMA)
// Block = 128 bucket rows of expert e. The gathered 128x256 bf16 token tile is
// staged into LDS once via global_load_async_to_lds_b128 (ASYNCcnt), so the hot
// loop's A-fragments come from LDS while the B-streams (W1t/W2t) flow through
// the vector-memory path. 8 waves; wave w owns rows [16w,16w+16) x 256 out
// channels (16 fp32 accum tiles). HID processed in 8 chunks of 64:
//   h = gelu(tok @ W1t^T + b1)  (bf16 via per-wave LDS)  ;  acc += h @ W2t^T.
__global__ __launch_bounds__(256) void k_moe(
    const unsigned short* __restrict__ tokB,   // [N_TOK][C]
    const unsigned short* __restrict__ w1t,    // [E][HID][C]
    const unsigned short* __restrict__ w2t,    // [E][C][HID]
    const float* __restrict__ b1,              // [E][HID]
    const float* __restrict__ b2,              // [E][C]
    const int*   __restrict__ btok,
    const float* __restrict__ bwt,
    const int*   __restrict__ counts,
    const float* __restrict__ scale,           // 1 elem
    float* __restrict__ out) {
  int e    = blockIdx.x >> 9;
  int tile = blockIdx.x & (MAXTILE - 1);
  int cnt  = counts[e];
  if (tile * 128 >= cnt) return;

  __shared__ int            s_tok[128];
  __shared__ float          s_w[128];
  __shared__ int            s_base[128];
  __shared__ unsigned short t_s[128 * TOK_STRIDE];      // 66 KiB token tile
  __shared__ unsigned short h_s[8 * 16 * H_STRIDE];     // 18 KiB gelu scratch

  int tid = threadIdx.x;
  if (tid < 128) {
    int idx   = tile * 128 + tid;
    int valid = idx < cnt;
    int t = valid ? btok[e * N_TOK + idx] : 0;
    s_tok[tid]  = t;
    s_w[tid]    = valid ? bwt[e * N_TOK + idx] : 0.0f;
    s_base[tid] = (t >> 12) * (C_DIM * HW_DIM) + (t & (HW_DIM - 1));
  }
  __syncthreads();

  // ---- async-stage the gathered token tile into LDS (2 threads per row)
  {
    int row  = tid >> 1;
    int part = tid & 1;                                  // 128 bf16 half-row
    const char* g = (const char*)(tokB + (size_t)s_tok[row] * C_DIM) + part * 256;
    unsigned l = (unsigned)(size_t)(void*)&t_s[row * TOK_STRIDE + part * 128];
#pragma unroll
    for (int j = 0; j < 16; ++j) {
      asm volatile("global_load_async_to_lds_b128 %0, %1, off"
                   :: "v"(l + j * 16), "v"(g + j * 16) : "memory");
    }
  }
  asm volatile("s_wait_asynccnt 0" ::: "memory");
  __syncthreads();

  int wave = tid >> 5, lane = tid & 31;
  int half = lane >> 4, ln = lane & 15;
  int m0 = wave * 16;

  const unsigned short* w1e = w1t + (size_t)e * HID_DIM * C_DIM;
  const unsigned short* w2e = w2t + (size_t)e * C_DIM * HID_DIM;
  const uint4* ta4 = (const uint4*)t_s;
  const uint4* hs4 = (const uint4*)h_s;
  int abase = (m0 + ln) * (TOK_STRIDE / 8);              // uint4 units (33/row)

  v8f acc2[16];
#pragma unroll
  for (int n = 0; n < 16; ++n)
#pragma unroll
    for (int j = 0; j < 8; ++j) acc2[n][j] = 0.0f;

#pragma unroll 1
  for (int chunk = 0; chunk < 8; ++chunk) {
    // ---- GEMM1: h[16 x 64] = tok[16 x 256] @ W1[:, chunk*64 + 0..63]
    v8f acc1[4];
#pragma unroll
    for (int n = 0; n < 4; ++n)
#pragma unroll
      for (int j = 0; j < 8; ++j) acc1[n][j] = 0.0f;

#pragma unroll 2
    for (int k = 0; k < 8; ++k) {              // K = 8 x 32 = 256
      Frag a;
      a.q[0] = ta4[abase + k * 4 + half];
      a.q[1] = ta4[abase + k * 4 + half + 2];
#pragma unroll
      for (int t4 = 0; t4 < 4; ++t4) {
        const uint4* bp =
            (const uint4*)(w1e + (size_t)(chunk * 64 + t4 * 16 + ln) * C_DIM);
        Frag b;
        b.q[0] = bp[k * 4 + half];
        b.q[1] = bp[k * 4 + half + 2];
        acc1[t4] = __builtin_amdgcn_wmma_f32_16x16x32_bf16(
            false, a.bf, false, b.bf, (short)0, acc1[t4], false, false);
      }
    }
    // ---- bias + exact gelu -> bf16 -> per-wave LDS (D-layout -> A-layout)
#pragma unroll
    for (int t4 = 0; t4 < 4; ++t4) {
      int hcol = chunk * 64 + t4 * 16 + ln;
      float b1v = b1[e * HID_DIM + hcol];
#pragma unroll
      for (int r = 0; r < 8; ++r) {
        float g = gelu_exact(acc1[t4][r] + b1v);
        h_s[wave * 16 * H_STRIDE + (r + 8 * half) * H_STRIDE + (t4 * 16 + ln)] =
            f2bf(g);
      }
    }
    asm volatile("s_wait_dscnt 0" ::: "memory");
    // ---- GEMM2 partial: acc2 += h[16 x 64] @ W2[chunk*64 + 0..63, :]
#pragma unroll
    for (int kc = 0; kc < 2; ++kc) {
      Frag ha;
      int hidx = wave * (16 * H_STRIDE / 8) + ln * (H_STRIDE / 8) + kc * 4 + half;
      ha.q[0] = hs4[hidx];
      ha.q[1] = hs4[hidx + 2];
#pragma unroll
      for (int n = 0; n < 16; ++n) {
        const uint4* bp = (const uint4*)(w2e + (size_t)(n * 16 + ln) * HID_DIM);
        Frag b;
        b.q[0] = bp[chunk * 8 + kc * 4 + half];
        b.q[1] = bp[chunk * 8 + kc * 4 + half + 2];
        acc2[n] = __builtin_amdgcn_wmma_f32_16x16x32_bf16(
            false, ha.bf, false, b.bf, (short)0, acc2[n], false, false);
      }
    }
  }

  // ---- epilogue: + b2, x (routing weight * scale), scatter-atomic into out
  float sc = scale[0];
  float b2v[16];
#pragma unroll
  for (int n = 0; n < 16; ++n) b2v[n] = b2[e * C_DIM + n * 16 + ln];
#pragma unroll
  for (int r = 0; r < 8; ++r) {
    int row = m0 + r + 8 * half;
    float wv  = s_w[row] * sc;
    int  base = s_base[row];
#pragma unroll
    for (int n = 0; n < 16; ++n) {
      int c = n * 16 + ln;
      unsafeAtomicAdd(out + base + (long)c * HW_DIM,
                      (acc2[n][r] + b2v[n]) * wv);
    }
  }
}

// ---------------------------------------------------------------------------
// Workspace layout (needs ~42 MB):
//   tok_bf16 [N][C]      : 33,554,432 B @ 0
//   W1t bf16 [E][HID][C] :  2,097,152 B @ 33,554,432
//   W2t bf16 [E][C][HID] :  2,097,152 B @ 35,651,584
//   bucket_tok int [E][N]:  2,097,152 B @ 37,748,736
//   bucket_w  f32 [E][N] :  2,097,152 B @ 39,845,888
//   cursors   int [8]    :         32 B @ 41,943,040
extern "C" void kernel_launch(void* const* d_in, const int* in_sizes, int n_in,
                              void* d_out, int out_size, void* d_ws, size_t ws_size,
                              hipStream_t stream) {
  const float* x     = (const float*)d_in[0];
  const float* Wr    = (const float*)d_in[1];
  const float* br    = (const float*)d_in[2];
  const float* W1    = (const float*)d_in[3];
  const float* b1    = (const float*)d_in[4];
  const float* W2    = (const float*)d_in[5];
  const float* b2    = (const float*)d_in[6];
  const float* scale = (const float*)d_in[7];
  float* out = (float*)d_out;

  char* ws = (char*)d_ws;
  unsigned short* tok = (unsigned short*)(ws);
  unsigned short* w1t = (unsigned short*)(ws + 33554432UL);
  unsigned short* w2t = (unsigned short*)(ws + 35651584UL);
  int*   btok    = (int*)  (ws + 37748736UL);
  float* bwt     = (float*)(ws + 39845888UL);
  int*   cursors = (int*)  (ws + 41943040UL);

  k_zero<<<1, 32, 0, stream>>>(cursors);
  k_copy<<<16384, 256, 0, stream>>>((const float4*)x, (float4*)out);

  // tokens: per batch, (C=256 x HW=4096) -> (HW x C) bf16
  k_transpose_bf16<<<dim3(HW_DIM / 32, C_DIM / 32, B_NUM), dim3(32, 8), 0, stream>>>(
      x, tok, C_DIM, HW_DIM, (long)C_DIM * HW_DIM, (long)C_DIM * HW_DIM);
  // W1: per expert, (C x HID) -> (HID x C) bf16
  k_transpose_bf16<<<dim3(HID_DIM / 32, C_DIM / 32, E_NUM), dim3(32, 8), 0, stream>>>(
      W1, w1t, C_DIM, HID_DIM, (long)C_DIM * HID_DIM, (long)C_DIM * HID_DIM);
  // W2: per expert, (HID x C) -> (C x HID) bf16
  k_transpose_bf16<<<dim3(C_DIM / 32, HID_DIM / 32, E_NUM), dim3(32, 8), 0, stream>>>(
      W2, w2t, HID_DIM, C_DIM, (long)C_DIM * HID_DIM, (long)C_DIM * HID_DIM);

  k_router<<<N_TOK / 256, 256, 0, stream>>>(tok, Wr, br, btok, bwt, cursors);

  k_moe<<<E_NUM * MAXTILE, 256, 0, stream>>>(tok, w1t, w2t, b1, b2,
                                             btok, bwt, cursors, scale, out);
}
